// AdditiveAttn_72610717106195
// MI455X (gfx1250) — matrix operand order repaired and checked by reading, not compiled, not run on hardware
//
#include <hip/hip_runtime.h>
#include <hip/hip_bf16.h>

typedef __attribute__((ext_vector_type(16))) __bf16 v16bf;
typedef __attribute__((ext_vector_type(8)))  __bf16 v8bf;
typedef __attribute__((ext_vector_type(8)))  float  v8f;

#define FDIM 512
#define HDIM 512
#define DIM  256
#define BATCH 64
#define NTOK 4096

#define ROWS 64                 // rows per block: 4 M-tiles -> 4x B-fragment reuse
#define MT   (ROWS / 16)
#define ASTRIDE (FDIM + 8)      // 520 ushorts/row -> 1040B stride, kills bank conflicts

#if __has_builtin(__builtin_amdgcn_tanhf)
  #define TANHF(x) __builtin_amdgcn_tanhf(x)
#elif __has_builtin(__builtin_amdgcn_tanh_f32)
  #define TANHF(x) __builtin_amdgcn_tanh_f32(x)
#else
  #define TANHF(x) tanhf(x)
#endif

__device__ __forceinline__ unsigned short f2bf(float x) {
    unsigned u = __float_as_uint(x);
    u += 0x7FFFu + ((u >> 16) & 1u);          // round-to-nearest-even
    return (unsigned short)(u >> 16);
}

// ---- W_img fp32 -> bf16 (once; 256KB stays L2-resident) ----
__global__ __launch_bounds__(256)
void wconv_kernel(const float* __restrict__ W, unsigned short* __restrict__ Wb) {
    int idx = blockIdx.x * 256 + threadIdx.x;   // DIM*FDIM = 131072
    Wb[idx] = f2bf(W[idx]);
}

// ---- proj_hidden[b][d] = hidden[b,:] . W_hid[d,:]  (tiny) ----
__global__ __launch_bounds__(256)
void ph_kernel(const float* __restrict__ hidden, const float* __restrict__ Whid,
               float* __restrict__ ph) {
    int b = blockIdx.x, d = threadIdx.x;        // blockDim = DIM = 256
    const float4* hv = (const float4*)(hidden + (size_t)b * HDIM);
    const float4* wv = (const float4*)(Whid   + (size_t)d * HDIM);
    float acc = 0.f;
    #pragma unroll 4
    for (int j = 0; j < HDIM / 4; ++j) {
        float4 h = hv[j], w = wv[j];
        acc += h.x * w.x + h.y * w.y + h.z * w.z + h.w * w.w;
    }
    ph[b * DIM + d] = acc;
}

// ---- fused: scores[r] = w_score . tanh(img[r,:] @ W_img^T + ph[b,:]) ----
// block = 64 rows (4 M-tiles); 8 waves x 2 D-tiles = full D=256
__global__ __launch_bounds__(256)
void scores_kernel(const float* __restrict__ img,
                   const unsigned short* __restrict__ Wb,   // [DIM][FDIM] bf16
                   const float* __restrict__ ph,            // [B][DIM]
                   const float* __restrict__ wscore,        // [DIM]
                   float* __restrict__ scores) {            // [B*N]
    __shared__ __align__(16) unsigned short Atile[ROWS][ASTRIDE];
    __shared__ float partials[ROWS][9];

    const int  tid     = threadIdx.x;
    const long rowbase = (long)blockIdx.x * ROWS;
    const int  b       = (int)(rowbase / NTOK);   // 4096 % 64 == 0 -> single b per block
    const float* Ab    = img + rowbase * FDIM;

    // stage 64x512 fp32 -> bf16 LDS (float4 loads, fully coalesced)
    const float4* Av = (const float4*)Ab;
    for (int it = tid; it < ROWS * FDIM / 4; it += 256) {
        float4 x = Av[it];
        int row = it >> 7;            // /128 float4 per row
        int col = (it & 127) * 4;
        unsigned short* dst = &Atile[row][col];
        dst[0] = f2bf(x.x); dst[1] = f2bf(x.y); dst[2] = f2bf(x.z); dst[3] = f2bf(x.w);
    }
    __syncthreads();

    const int wave = tid >> 5;
    const int lane = tid & 31;
    const int half = lane >> 4;     // K-half select: k = 16*half + s (contiguous order)
    const int m16  = lane & 15;

    union Frag { v16bf v; v8bf h[2]; };

    const int d0 = (wave * 2)     * 16 + m16;
    const int d1 = (wave * 2 + 1) * 16 + m16;
    const unsigned short* B0 = Wb + (size_t)d0 * FDIM;
    const unsigned short* B1 = Wb + (size_t)d1 * FDIM;

    v8f c0[MT], c1[MT];
    #pragma unroll
    for (int mt = 0; mt < MT; ++mt) { c0[mt] = (v8f){}; c1[mt] = (v8f){}; }

    #pragma unroll 2
    for (int kt = 0; kt < FDIM / 32; ++kt) {
        // B 32x16: lane = column d; lanes 0-15 K 0..15, lanes 16-31 K 16..31
        const int kB = kt * 32 + half * 16;
        Frag fb0, fb1;
        fb0.h[0] = *(const v8bf*)&B0[kB];
        fb0.h[1] = *(const v8bf*)&B0[kB + 8];
        fb1.h[0] = *(const v8bf*)&B1[kB];
        fb1.h[1] = *(const v8bf*)&B1[kB + 8];
        // A 16x32: lanes 0-15 hold K 0..15, lanes 16-31 hold K 16..31
        const int kA = kt * 32 + half * 16;
        #pragma unroll
        for (int mt = 0; mt < MT; ++mt) {
            Frag a;
            a.h[0] = *(const v8bf*)&Atile[mt * 16 + m16][kA];
            a.h[1] = *(const v8bf*)&Atile[mt * 16 + m16][kA + 8];
            c0[mt] = __builtin_amdgcn_wmma_f32_16x16x32_bf16(false, a.v, false, fb0.v,
                                                             (short)0, c0[mt], false, false);
            c1[mt] = __builtin_amdgcn_wmma_f32_16x16x32_bf16(false, a.v, false, fb1.v,
                                                             (short)0, c1[mt], false, false);
        }
    }

    // fused epilogue: tanh(+ph) . w_score, reduce over this wave's 32 D-columns
    const float ph0 = ph[b * DIM + d0], ws0 = wscore[d0];
    const float ph1 = ph[b * DIM + d1], ws1 = wscore[d1];
    #pragma unroll
    for (int mt = 0; mt < MT; ++mt) {
        #pragma unroll
        for (int i = 0; i < 8; ++i) {
            // C layout: VGPR i -> row m = i + half*8, col = lane&15
            float s = TANHF(c0[mt][i] + ph0) * ws0 + TANHF(c1[mt][i] + ph1) * ws1;
            s += __shfl_xor(s, 1, 32);
            s += __shfl_xor(s, 2, 32);
            s += __shfl_xor(s, 4, 32);
            s += __shfl_xor(s, 8, 32);      // reduce within 16-lane half (row fixed)
            if (m16 == 0) partials[mt * 16 + i + half * 8][wave] = s;
        }
    }
    __syncthreads();
    if (tid < ROWS) {
        float s = 0.f;
        #pragma unroll
        for (int w = 0; w < 8; ++w) s += partials[tid][w];
        scores[rowbase + tid] = s;
    }
}

// ---- softmax over N=4096 per batch row ----
__global__ __launch_bounds__(256)
void softmax_kernel(const float* __restrict__ scores, float* __restrict__ weights) {
    __shared__ float red[8];
    const int b = blockIdx.x, tid = threadIdx.x;
    const float* s = scores + (size_t)b * NTOK;
    float loc[16];
    float mx = -3.402823466e38f;
    #pragma unroll
    for (int i = 0; i < 16; ++i) { loc[i] = s[tid + i * 256]; mx = fmaxf(mx, loc[i]); }
    for (int m = 1; m < 32; m <<= 1) mx = fmaxf(mx, __shfl_xor(mx, m, 32));
    if ((tid & 31) == 0) red[tid >> 5] = mx;
    __syncthreads();
    mx = red[0];
    #pragma unroll
    for (int w = 1; w < 8; ++w) mx = fmaxf(mx, red[w]);
    __syncthreads();
    float sum = 0.f;
    #pragma unroll
    for (int i = 0; i < 16; ++i) { loc[i] = __expf(loc[i] - mx); sum += loc[i]; }
    for (int m = 1; m < 32; m <<= 1) sum += __shfl_xor(sum, m, 32);
    if ((tid & 31) == 0) red[tid >> 5] = sum;
    __syncthreads();
    sum = 0.f;
    #pragma unroll
    for (int w = 0; w < 8; ++w) sum += red[w];
    const float inv = 1.0f / sum;
    float* wout = weights + (size_t)b * NTOK;
    #pragma unroll
    for (int i = 0; i < 16; ++i) wout[tid + i * 256] = loc[i] * inv;
}

// ---- context partials: deterministic split over 16 N-chunks (no FP atomics) ----
__global__ __launch_bounds__(128)
void ctx_partial_kernel(const float* __restrict__ img, const float* __restrict__ weights,
                        float* __restrict__ cpart) {
    const int chunk = blockIdx.x;     // 0..15, 256 tokens each
    const int b     = blockIdx.y;
    const int tid   = threadIdx.x;    // 0..127 -> float4 over F=512
    const int n0    = chunk * 256;
    const float4* iv = (const float4*)(img + ((size_t)b * NTOK + n0) * FDIM);
    const float*  wv = weights + (size_t)b * NTOK + n0;
    float4 acc = make_float4(0.f, 0.f, 0.f, 0.f);
    for (int n = 0; n < 256; ++n) {
        float w  = wv[n];
        float4 x = iv[(size_t)n * 128 + tid];
        acc.x += w * x.x; acc.y += w * x.y; acc.z += w * x.z; acc.w += w * x.w;
    }
    ((float4*)(cpart + (size_t)(chunk * BATCH + b) * FDIM))[tid] = acc;
}

__global__ __launch_bounds__(256)
void ctx_reduce_kernel(const float* __restrict__ cpart, float* __restrict__ ctx) {
    const int idx = blockIdx.x * 256 + threadIdx.x;   // < BATCH*FDIM = 32768
    float s = 0.f;
    #pragma unroll
    for (int c = 0; c < 16; ++c) s += cpart[(size_t)c * BATCH * FDIM + idx];
    ctx[idx] = s;
}

extern "C" void kernel_launch(void* const* d_in, const int* in_sizes, int n_in,
                              void* d_out, int out_size, void* d_ws, size_t ws_size,
                              hipStream_t stream) {
    const float* img    = (const float*)d_in[0];   // [64,4096,512]
    const float* hidden = (const float*)d_in[1];   // [64,512]
    const float* Wimg   = (const float*)d_in[2];   // [256,512]
    const float* Whid   = (const float*)d_in[3];   // [256,512]
    const float* Wscore = (const float*)d_in[4];   // [1,256]

    float* ctx_out = (float*)d_out;                  // context: 64*512
    float* w_out   = (float*)d_out + BATCH * FDIM;   // weights: 64*4096

    char* ws = (char*)d_ws;
    float*          ph_ws  = (float*)(ws);                               //  64 KB
    unsigned short* Wb     = (unsigned short*)(ws + 65536);              // 256 KB
    float*          sc_ws  = (float*)(ws + 65536 + 262144);              //   1 MB
    float*          cpart  = (float*)(ws + 65536 + 262144 + 1048576);    //   2 MB

    wconv_kernel<<<DIM * FDIM / 256, 256, 0, stream>>>(Wimg, Wb);
    ph_kernel<<<BATCH, DIM, 0, stream>>>(hidden, Whid, ph_ws);
    scores_kernel<<<BATCH * NTOK / ROWS, 256, 0, stream>>>(img, Wb, ph_ws, Wscore, sc_ws);
    softmax_kernel<<<BATCH, 256, 0, stream>>>(sc_ws, w_out);
    ctx_partial_kernel<<<dim3(16, BATCH), 128, 0, stream>>>(img, w_out, cpart);
    ctx_reduce_kernel<<<BATCH * FDIM / 256, 256, 0, stream>>>(cpart, ctx_out);
}
